// LinearAttention_36747740185038
// MI455X (gfx1250) — matrix operand order repaired and checked by reading, not compiled, not run on hardware
//
#include <hip/hip_runtime.h>
#include <stdint.h>

// Linear attention (fast-transformers, elu+1 feature map) for gfx1250 / MI455X.
// B=8, S=4096, D=512, H=8, DH=64. All GEMMs via v_wmma_f32_16x16x32_bf16.
// Bandwidth-bound (~73 GFLOP vs ~250 MB HBM traffic); intermediates kept bf16.

#define Bn 8
#define Sn 4096
#define Dn 512
#define Hn 8
#define DHn 64
#define EPSc 1e-6f

typedef __attribute__((ext_vector_type(16))) __bf16 v16bf;
typedef __attribute__((ext_vector_type(8)))  __bf16 v8bf;
typedef __attribute__((ext_vector_type(8)))  float  v8f;

union Frag {
  v16bf v;
  v8bf  h[2];
  __bf16 e[16];
};

static __device__ __forceinline__ unsigned short bfbits(__bf16 h) {
  union { __bf16 h; unsigned short s; } c; c.h = h; return c.s;
}
static __device__ __forceinline__ unsigned short f2bfbits(float f) {
  return bfbits((__bf16)f);               // native v_cvt to bf16
}
static __device__ __forceinline__ float bf2f(unsigned short s) {
  union { uint32_t u; float f; } c; c.u = ((uint32_t)s) << 16; return c.f;
}
static __device__ __forceinline__ float phi_map(float x) { // elu(x)+1
  return x > 0.0f ? x + 1.0f : __expf(x);
}

// A fragment from f32 memory (contiguous K order: element s of lane half h is k = 16h + s).
static __device__ __forceinline__ void loadA_f32(Frag& a, const float* xr, int k0, int kA1) {
#pragma unroll
  for (int i = 0; i < 4; ++i) {
    float2 f0 = *(const float2*)(xr + k0 + kA1 + 2 * i);
    float2 f1 = *(const float2*)(xr + k0 + kA1 + 8 + 2 * i);
    a.e[2 * i]     = (__bf16)f0.x;
    a.e[2 * i + 1] = (__bf16)f0.y;
    a.e[8 + 2 * i] = (__bf16)f1.x;
    a.e[9 + 2 * i] = (__bf16)f1.y;
  }
}
// A fragment from bf16 memory (two 16B vector loads).
static __device__ __forceinline__ void loadA_bf(Frag& a, const unsigned short* ar, int k0, int kA1) {
  a.h[0] = *(const v8bf*)(ar + k0 + kA1);
  a.h[1] = *(const v8bf*)(ar + k0 + kA1 + 8);
}

// ---------------------------------------------------------------- utilities
__global__ void k_zero(float* __restrict__ p, int n) {
  int i = blockIdx.x * 256 + threadIdx.x;
  if (i < n) p[i] = 0.0f;
}

// Convert + transpose weights to bf16: WT[mat][n*512 + k] = bf16(W[mat][k*512 + n]).
// Transposed layout makes WMMA B-fragments contiguous 32B per-lane loads.
__global__ __launch_bounds__(256) void k_convw(
    const float* __restrict__ Wq, const float* __restrict__ Wk,
    const float* __restrict__ Wv, const float* __restrict__ Wo,
    unsigned short* __restrict__ WTall)
{
  const int mat = blockIdx.y;
  const float* W = (mat == 0) ? Wq : (mat == 1) ? Wk : (mat == 2) ? Wv : Wo;
  int i = blockIdx.x * 256 + threadIdx.x;        // 0..262143
  int k = i >> 9, n = i & 511;
  WTall[(size_t)mat * 262144 + (size_t)n * 512 + k] = f2bfbits(W[(size_t)k * 512 + n]);
}

// ---------------------------------------------------------------- QKV projection
// grid (256 row-blocks of 128, 8 col-tiles of 64, 3 matrices), block 256 (8 waves).
// Each wave: 16 rows x 64 cols = 4 D-fragments, K loop 512 step 32, software-pipelined.
__global__ __launch_bounds__(256) void k_qkv(
    const float* __restrict__ x,
    const unsigned short* __restrict__ WTall,
    const float* __restrict__ bq, const float* __restrict__ bk, const float* __restrict__ bv,
    unsigned short* __restrict__ Qf, unsigned short* __restrict__ Kf,
    unsigned short* __restrict__ Vb)
{
  const int mat = blockIdx.z;
  const unsigned short* WT = WTall + (size_t)mat * 262144;
  const float* bias = (mat == 0) ? bq : (mat == 1) ? bk : bv;
  unsigned short* outb = (mat == 0) ? Qf : (mat == 1) ? Kf : Vb;

  const int lane = threadIdx.x & 31;
  const int wave = threadIdx.x >> 5;
  const bool hi  = lane >= 16;
  const int m    = lane & 15;
  const int rowBase = blockIdx.x * 128 + wave * 16;
  const int colBase = blockIdx.y * 64;
  const int kA1 = hi ? 16 : 0;
  const int kbb = hi ? 16 : 0;

  const float* xr = x + (size_t)(rowBase + m) * Dn;
  const unsigned short* w0 = WT + (size_t)(colBase +  0 + m) * 512 + kbb;
  const unsigned short* w1 = WT + (size_t)(colBase + 16 + m) * 512 + kbb;
  const unsigned short* w2 = WT + (size_t)(colBase + 32 + m) * 512 + kbb;
  const unsigned short* w3 = WT + (size_t)(colBase + 48 + m) * 512 + kbb;

  v8f c0 = {}, c1 = {}, c2 = {}, c3 = {};

  // software pipeline: preload next K-slice before consuming current one
  Frag a;  v16bf b0, b1, b2, b3;
  loadA_f32(a, xr, 0, kA1);
  b0 = *(const v16bf*)(w0);
  b1 = *(const v16bf*)(w1);
  b2 = *(const v16bf*)(w2);
  b3 = *(const v16bf*)(w3);

  for (int k0 = 0; k0 < Dn - 32; k0 += 32) {
    const int kn = k0 + 32;
    __builtin_prefetch(xr + kn + 32, 0, 3);      // global_prefetch_b8
    Frag an;  v16bf n0, n1, n2, n3;
    loadA_f32(an, xr, kn, kA1);
    n0 = *(const v16bf*)(w0 + kn);
    n1 = *(const v16bf*)(w1 + kn);
    n2 = *(const v16bf*)(w2 + kn);
    n3 = *(const v16bf*)(w3 + kn);
    c0 = __builtin_amdgcn_wmma_f32_16x16x32_bf16(false, a.v, false, b0, (short)0, c0, false, false);
    c1 = __builtin_amdgcn_wmma_f32_16x16x32_bf16(false, a.v, false, b1, (short)0, c1, false, false);
    c2 = __builtin_amdgcn_wmma_f32_16x16x32_bf16(false, a.v, false, b2, (short)0, c2, false, false);
    c3 = __builtin_amdgcn_wmma_f32_16x16x32_bf16(false, a.v, false, b3, (short)0, c3, false, false);
    a = an;  b0 = n0;  b1 = n1;  b2 = n2;  b3 = n3;
  }
  c0 = __builtin_amdgcn_wmma_f32_16x16x32_bf16(false, a.v, false, b0, (short)0, c0, false, false);
  c1 = __builtin_amdgcn_wmma_f32_16x16x32_bf16(false, a.v, false, b1, (short)0, c1, false, false);
  c2 = __builtin_amdgcn_wmma_f32_16x16x32_bf16(false, a.v, false, b2, (short)0, c2, false, false);
  c3 = __builtin_amdgcn_wmma_f32_16x16x32_bf16(false, a.v, false, b3, (short)0, c3, false, false);

  // Epilogue: bias, phi (Q/K only), store bf16. D layout: VGPR g -> row g / g+8.
#pragma unroll
  for (int f = 0; f < 4; ++f) {
    const v8f cc = (f == 0) ? c0 : (f == 1) ? c1 : (f == 2) ? c2 : c3;
    const int col = colBase + f * 16 + m;
    const float bvl = bias[col];
#pragma unroll
    for (int g = 0; g < 8; ++g) {
      const int r = rowBase + (hi ? g + 8 : g);
      float val = cc[g] + bvl;
      if (mat < 2) val = phi_map(val);
      outb[(size_t)r * Dn + col] = f2bfbits(val);
    }
  }
}

// ---------------------------------------------------------------- KV accumulation
// KV[b,h,d,m] = sum_s Kf[b,s,h,d]*V[b,s,h,m]; Ksum[b,h,d] = sum_s Kf.
// Tiny (2 GFLOP): split-S VALU reduction with f32 atomics. grid (64 bh, 16 splits).
__global__ __launch_bounds__(256) void k_kv(
    const unsigned short* __restrict__ Kf, const unsigned short* __restrict__ Vb,
    float* __restrict__ KV, float* __restrict__ Ksum)
{
  __shared__ unsigned short lk[8][64];
  __shared__ unsigned short lv[8][64];
  const int bh = blockIdx.x, b = bh >> 3, h = bh & 7;
  const int s0 = blockIdx.y * 256;
  const int t = threadIdx.x;
  const int d = t >> 2;
  const int m0 = (t & 3) * 16;
  float acc[16] = {};
  float ksum = 0.0f;

  for (int so = 0; so < 256; so += 8) {
    __syncthreads();
    for (int i = t; i < 512; i += 256) {
      int ss = i >> 6, e = i & 63;
      size_t g = ((size_t)b * Sn + s0 + so + ss) * Dn + h * DHn + e;
      lk[ss][e] = Kf[g];
      lv[ss][e] = Vb[g];
    }
    __syncthreads();
#pragma unroll
    for (int ss = 0; ss < 8; ++ss) {
      float kd = bf2f(lk[ss][d]);
      if ((t & 3) == 0) ksum += kd;
#pragma unroll
      for (int j = 0; j < 16; ++j) acc[j] += kd * bf2f(lv[ss][m0 + j]);
    }
  }
  float* kvp = KV + (size_t)bh * 4096 + d * 64 + m0;
#pragma unroll
  for (int j = 0; j < 16; ++j) atomicAdd(kvp + j, acc[j]);
  if ((t & 3) == 0) atomicAdd(Ksum + (size_t)bh * 64 + d, ksum);
}

// ---------------------------------------------------------------- Qf @ KV * Z
// grid (64 bh, 32 s-blocks of 128), block 256. KV -> bf16 transposed in LDS.
__global__ __launch_bounds__(256) void k_attn(
    const unsigned short* __restrict__ Qf,
    const float* __restrict__ KV, const float* __restrict__ Ksum,
    unsigned short* __restrict__ attn)
{
  __shared__ unsigned short kvT[64 * 64]; // kvT[m*64 + d]
  __shared__ float ksl[64];
  const int bh = blockIdx.x, b = bh >> 3, h = bh & 7;
  const float* kvp = KV + (size_t)bh * 4096;
  for (int i = threadIdx.x; i < 4096; i += 256) {
    int d = i >> 6, mm = i & 63;
    kvT[mm * 64 + d] = f2bfbits(kvp[i]);
  }
  if (threadIdx.x < 64) ksl[threadIdx.x] = Ksum[(size_t)bh * 64 + threadIdx.x];
  __syncthreads();

  const int lane = threadIdx.x & 31;
  const int wave = threadIdx.x >> 5;
  const bool hi = lane >= 16;
  const int m = lane & 15;
  const int sBase = blockIdx.y * 128 + wave * 16;
  const int kA1 = hi ? 16 : 0;
  const size_t qbase = ((size_t)b * Sn + sBase) * Dn + h * DHn;

  v8f c0 = {}, c1 = {}, c2 = {}, c3 = {};
#pragma unroll
  for (int k0 = 0; k0 < 64; k0 += 32) {
    Frag a;
    loadA_bf(a, Qf + qbase + (size_t)m * Dn, k0, kA1);
    const int kb = k0 + (hi ? 16 : 0);
    v16bf b0 = *(const v16bf*)&kvT[( 0 + m) * 64 + kb];
    v16bf b1 = *(const v16bf*)&kvT[(16 + m) * 64 + kb];
    v16bf b2 = *(const v16bf*)&kvT[(32 + m) * 64 + kb];
    v16bf b3 = *(const v16bf*)&kvT[(48 + m) * 64 + kb];
    c0 = __builtin_amdgcn_wmma_f32_16x16x32_bf16(false, a.v, false, b0, (short)0, c0, false, false);
    c1 = __builtin_amdgcn_wmma_f32_16x16x32_bf16(false, a.v, false, b1, (short)0, c1, false, false);
    c2 = __builtin_amdgcn_wmma_f32_16x16x32_bf16(false, a.v, false, b2, (short)0, c2, false, false);
    c3 = __builtin_amdgcn_wmma_f32_16x16x32_bf16(false, a.v, false, b3, (short)0, c3, false, false);
  }

  // Z[row] = 1/(Qf[row,:].Ksum + eps); every lane computes row = lane&15 (no divergence).
  float zacc = EPSc;
  const unsigned short* qz = Qf + ((size_t)b * Sn + sBase + m) * Dn + h * DHn;
#pragma unroll
  for (int i = 0; i < 8; ++i) {
    v8bf qv = *(const v8bf*)(qz + 8 * i);
#pragma unroll
    for (int j = 0; j < 8; ++j) zacc += (float)qv[j] * ksl[8 * i + j];
  }
  float zv = 1.0f / zacc;

#pragma unroll
  for (int g = 0; g < 8; ++g) {
    const int M = hi ? g + 8 : g;
    const float zr = __shfl(zv, M, 32);   // rows 0..15 live in lanes 0..15
    const size_t rb = ((size_t)b * Sn + sBase + M) * Dn + h * DHn;
    attn[rb +  0 + m] = f2bfbits(c0[g] * zr);
    attn[rb + 16 + m] = f2bfbits(c1[g] * zr);
    attn[rb + 32 + m] = f2bfbits(c2[g] * zr);
    attn[rb + 48 + m] = f2bfbits(c3[g] * zr);
  }
}

// ---------------------------------------------------------------- output projection
// grid (256 row-blocks, 8 col-tiles), block 256; software-pipelined K loop.
__global__ __launch_bounds__(256) void k_oproj(
    const unsigned short* __restrict__ attn,
    const unsigned short* __restrict__ WoT,
    const float* __restrict__ bo,
    float* __restrict__ out)
{
  const int lane = threadIdx.x & 31;
  const int wave = threadIdx.x >> 5;
  const bool hi = lane >= 16;
  const int m = lane & 15;
  const int rowBase = blockIdx.x * 128 + wave * 16;
  const int colBase = blockIdx.y * 64;
  const int kA1 = hi ? 16 : 0;
  const int kbb = hi ? 16 : 0;
  const unsigned short* ar = attn + (size_t)(rowBase + m) * Dn;
  const unsigned short* w0 = WoT + (size_t)(colBase +  0 + m) * 512 + kbb;
  const unsigned short* w1 = WoT + (size_t)(colBase + 16 + m) * 512 + kbb;
  const unsigned short* w2 = WoT + (size_t)(colBase + 32 + m) * 512 + kbb;
  const unsigned short* w3 = WoT + (size_t)(colBase + 48 + m) * 512 + kbb;

  v8f c0 = {}, c1 = {}, c2 = {}, c3 = {};

  Frag a;  v16bf b0, b1, b2, b3;
  loadA_bf(a, ar, 0, kA1);
  b0 = *(const v16bf*)(w0);
  b1 = *(const v16bf*)(w1);
  b2 = *(const v16bf*)(w2);
  b3 = *(const v16bf*)(w3);

  for (int k0 = 0; k0 < Dn - 32; k0 += 32) {
    const int kn = k0 + 32;
    __builtin_prefetch(ar + kn + 32, 0, 3);
    Frag an;  v16bf n0, n1, n2, n3;
    loadA_bf(an, ar, kn, kA1);
    n0 = *(const v16bf*)(w0 + kn);
    n1 = *(const v16bf*)(w1 + kn);
    n2 = *(const v16bf*)(w2 + kn);
    n3 = *(const v16bf*)(w3 + kn);
    c0 = __builtin_amdgcn_wmma_f32_16x16x32_bf16(false, a.v, false, b0, (short)0, c0, false, false);
    c1 = __builtin_amdgcn_wmma_f32_16x16x32_bf16(false, a.v, false, b1, (short)0, c1, false, false);
    c2 = __builtin_amdgcn_wmma_f32_16x16x32_bf16(false, a.v, false, b2, (short)0, c2, false, false);
    c3 = __builtin_amdgcn_wmma_f32_16x16x32_bf16(false, a.v, false, b3, (short)0, c3, false, false);
    a = an;  b0 = n0;  b1 = n1;  b2 = n2;  b3 = n3;
  }
  c0 = __builtin_amdgcn_wmma_f32_16x16x32_bf16(false, a.v, false, b0, (short)0, c0, false, false);
  c1 = __builtin_amdgcn_wmma_f32_16x16x32_bf16(false, a.v, false, b1, (short)0, c1, false, false);
  c2 = __builtin_amdgcn_wmma_f32_16x16x32_bf16(false, a.v, false, b2, (short)0, c2, false, false);
  c3 = __builtin_amdgcn_wmma_f32_16x16x32_bf16(false, a.v, false, b3, (short)0, c3, false, false);

#pragma unroll
  for (int f = 0; f < 4; ++f) {
    const v8f cc = (f == 0) ? c0 : (f == 1) ? c1 : (f == 2) ? c2 : c3;
    const int col = colBase + f * 16 + m;
    const float bvl = bo[col];
#pragma unroll
    for (int g = 0; g < 8; ++g) {
      const int r = rowBase + (hi ? g + 8 : g);
      out[(size_t)r * Dn + col] = cc[g] + bvl;
    }
  }
}

// ---------------------------------------------------------------- launch
extern "C" void kernel_launch(void* const* d_in, const int* in_sizes, int n_in,
                              void* d_out, int out_size, void* d_ws, size_t ws_size,
                              hipStream_t stream)
{
  (void)in_sizes; (void)n_in; (void)out_size; (void)ws_size;
  const float* x  = (const float*)d_in[0];
  const float* Wq = (const float*)d_in[1];
  const float* bq = (const float*)d_in[2];
  const float* Wk = (const float*)d_in[3];
  const float* bk = (const float*)d_in[4];
  const float* Wv = (const float*)d_in[5];
  const float* bv = (const float*)d_in[6];
  const float* Wo = (const float*)d_in[7];
  const float* bo = (const float*)d_in[8];
  float* out = (float*)d_out;
  char* ws = (char*)d_ws;

  // Workspace layout (132 MB total):
  unsigned short* WTall = (unsigned short*)(ws);                       // 2 MB  (q,k,v,o transposed bf16)
  float*          KV    = (float*)(ws + (size_t)(2)   * (1u << 20));   // 1 MB
  unsigned short* Qf    = (unsigned short*)(ws + (size_t)(4)   * (1u << 20)); // 32 MB
  unsigned short* Kf    = (unsigned short*)(ws + (size_t)(36)  * (1u << 20)); // 32 MB
  unsigned short* Vb    = (unsigned short*)(ws + (size_t)(68)  * (1u << 20)); // 32 MB
  unsigned short* attn  = (unsigned short*)(ws + (size_t)(100) * (1u << 20)); // 32 MB
  float*          Ksum  = (float*)(ws + (size_t)(3)   * (1u << 20));   // 16 KB (contiguous after KV)

  // 1) zero KV + Ksum (contiguous region)
  {
    int n = 262144 + 4096;
    k_zero<<<dim3((n + 255) / 256), dim3(256), 0, stream>>>(KV, n);
  }
  // 2) weight convert+transpose to bf16
  k_convw<<<dim3(1024, 4), dim3(256), 0, stream>>>(Wq, Wk, Wv, Wo, WTall);
  // 3) QKV projections + phi (WMMA)
  k_qkv<<<dim3(256, 8, 3), dim3(256), 0, stream>>>(x, WTall, bq, bk, bv, Qf, Kf, Vb);
  // 4) KV / Ksum reduction
  k_kv<<<dim3(64, 16), dim3(256), 0, stream>>>(Kf, Vb, KV, Ksum);
  // 5) Qf @ KV scaled by Z (WMMA)
  k_attn<<<dim3(64, 32), dim3(256), 0, stream>>>(Qf, KV, Ksum, attn);
  // 6) output projection (WMMA)
  k_oproj<<<dim3(256, 8), dim3(256), 0, stream>>>(attn, WTall + (size_t)3 * 262144, bo, out);
}
